// PointRend_36541581754598
// MI455X (gfx1250) — matrix operand order repaired and checked by reading, not compiled, not run on hardware
//
#include <hip/hip_runtime.h>
#include <hip/hip_bf16.h>

// ---------------- problem constants ----------------
#define NB       4
#define CO       128          // coarse H=W
#define FI       256          // fine H=W
#define FIC      256          // fine channels
#define NCLS     19
#define PTR      8192         // P_TRAIN
#define OSP      24576        // P_TRAIN*OVERSAMPLE
#define NPAD     32768        // sort pad (pow2 >= OSP)
#define NUNC     6144
#define NRAND    2048
#define KF       288          // padded feature dim (275 -> 288 = 9*32)
#define HID      256
#define NPTS     (NB*PTR)     // 32768 total points

// d_out layout (floats): probs | point_logits | point_coords
#define OUT_LOGITS  ((size_t)NB*512*512*NCLS)                 // 19,922,944
#define OUT_COORDS  (OUT_LOGITS + (size_t)NB*PTR*NCLS)        // 20,545,536

// workspace byte offsets
#define WS_KEYS   ((size_t)0)                                 // u64 [NB][NPAD]     = 1 MiB
#define WS_WF0    ((size_t)(1u<<20))
#define WS_WF1    (WS_WF0 + (size_t)9*16*32*16*2)             // 147456 B each hidden layer
#define WS_WF2    (WS_WF1 + (size_t)9*16*32*16*2)
#define WS_WFF    (WS_WF2 + (size_t)9*16*32*16*2)             // final: 9*2*32*16*2 = 18432 B
#define WS_FEAT0  ((size_t)1572864)                           // f16 [NPTS][KF] = 18,874,368 B
#define WS_X      (WS_FEAT0 + (size_t)NPTS*KF*2)              // f16 [NPTS][KF]

typedef __attribute__((ext_vector_type(16))) _Float16 v16h;
typedef __attribute__((ext_vector_type(8)))  float    v8f;

// monotone key: larger float -> larger unsigned
__device__ __forceinline__ unsigned fkey(float f) {
  unsigned u = __float_as_uint(f);
  return (u & 0x80000000u) ? ~u : (u | 0x80000000u);
}

// ---------------- 1) uncertainty keys ----------------
__global__ void __launch_bounds__(256) k_uncertainty(const float* __restrict__ coarse,
                                                     const float* __restrict__ rc,
                                                     unsigned long long* __restrict__ keys) {
  int i = blockIdx.x * blockDim.x + threadIdx.x;
  if (i >= NB * NPAD) return;
  int b = i >> 15, j = i & (NPAD - 1);
  unsigned long long key = 0ull;
  if (j < OSP) {
    float cy = rc[((size_t)b * OSP + j) * 2 + 0];
    float cx = rc[((size_t)b * OSP + j) * 2 + 1];
    float y = cy * (float)CO - 0.5f, x = cx * (float)CO - 0.5f;
    float y0f = floorf(y), x0f = floorf(x);
    float wy = y - y0f, wx = x - x0f;
    int y0 = min(max((int)y0f, 0), CO - 1), y1 = min(max((int)y0f + 1, 0), CO - 1);
    int x0 = min(max((int)x0f, 0), CO - 1), x1 = min(max((int)x0f + 1, 0), CO - 1);
    const float* f00 = coarse + (((size_t)b * CO + y0) * CO + x0) * NCLS;
    const float* f01 = coarse + (((size_t)b * CO + y0) * CO + x1) * NCLS;
    const float* f10 = coarse + (((size_t)b * CO + y1) * CO + x0) * NCLS;
    const float* f11 = coarse + (((size_t)b * CO + y1) * CO + x1) * NCLS;
    float m1 = -3.4e38f, m2 = -3.4e38f;
#pragma unroll
    for (int c = 0; c < NCLS; ++c) {
      float v = (1.f - wy) * ((1.f - wx) * f00[c] + wx * f01[c]) +
                wy         * ((1.f - wx) * f10[c] + wx * f11[c]);
      if (v > m1) { m2 = m1; m1 = v; } else if (v > m2) { m2 = v; }
    }
    float unc = m2 - m1;                       // higher = more uncertain
    key = ((unsigned long long)fkey(unc) << 32) | (unsigned)(~(unsigned)j);
  }
  keys[(size_t)b * NPAD + j] = key;
}

// ---------------- 2) global bitonic sort step (descending) ----------------
__global__ void __launch_bounds__(256) k_sortstep(unsigned long long* __restrict__ keys, int j, int k) {
  int i = blockIdx.x * blockDim.x + threadIdx.x;
  if (i >= NB * NPAD) return;
  int b = i >> 15, t = i & (NPAD - 1);
  int ixj = t ^ j;
  if (ixj > t) {
    unsigned long long* base = keys + ((size_t)b << 15);
    unsigned long long a = base[t], c = base[ixj];
    bool up = ((t & k) == 0);                  // descending blocks -> global descending
    bool sw = up ? (a < c) : (a > c);
    if (sw) { base[t] = c; base[ixj] = a; }
  }
}

// ---------------- 3) gather point_coords ----------------
__global__ void __launch_bounds__(256) k_gather(const unsigned long long* __restrict__ keys,
                                                const float* __restrict__ rc,
                                                const float* __restrict__ rf,
                                                float* __restrict__ outc) {
  int i = blockIdx.x * blockDim.x + threadIdx.x;
  if (i >= NPTS) return;
  int b = i >> 13, p = i & (PTR - 1);
  float cy, cx;
  if (p < NUNC) {
    unsigned idx = ~(unsigned)(keys[(size_t)b * NPAD + p]);
    cy = rc[((size_t)b * OSP + idx) * 2 + 0];
    cx = rc[((size_t)b * OSP + idx) * 2 + 1];
  } else {
    int q = p - NUNC;
    cy = rf[((size_t)b * NRAND + q) * 2 + 0];
    cx = rf[((size_t)b * NRAND + q) * 2 + 1];
  }
  outc[(size_t)i * 2 + 0] = cy;
  outc[(size_t)i * 2 + 1] = cx;
}

// ---------------- 4) sample features -> feat0 [coarse|fine|0], X pad+coarse cols ----------------
__global__ void __launch_bounds__(256) k_sample(const float* __restrict__ coarse,
                                                const float* __restrict__ fine,
                                                const float* __restrict__ outc,
                                                _Float16* __restrict__ feat0,
                                                _Float16* __restrict__ X) {
  int pt   = blockIdx.x * 8 + (threadIdx.x >> 5);   // 8 points / block, 1 wave / point
  int lane = threadIdx.x & 31;
  int b = pt >> 13;
  float cy = outc[(size_t)pt * 2 + 0];
  float cx = outc[(size_t)pt * 2 + 1];

  // fine grid 256x256
  float yf = cy * (float)FI - 0.5f, xf = cx * (float)FI - 0.5f;
  float yf0f = floorf(yf), xf0f = floorf(xf);
  float wyf = yf - yf0f, wxf = xf - xf0f;
  int fy0 = min(max((int)yf0f, 0), FI - 1), fy1 = min(max((int)yf0f + 1, 0), FI - 1);
  int fx0 = min(max((int)xf0f, 0), FI - 1), fx1 = min(max((int)xf0f + 1, 0), FI - 1);
  size_t f00 = (((size_t)b * FI + fy0) * FI + fx0) * FIC;
  size_t f01 = (((size_t)b * FI + fy0) * FI + fx1) * FIC;
  size_t f10 = (((size_t)b * FI + fy1) * FI + fx0) * FIC;
  size_t f11 = (((size_t)b * FI + fy1) * FI + fx1) * FIC;

  _Float16* frow = feat0 + (size_t)pt * KF;
#pragma unroll
  for (int j = 0; j < 8; ++j) {
    int c = lane + 32 * j;                          // coalesced across lanes
    float v = (1.f - wyf) * ((1.f - wxf) * fine[f00 + c] + wxf * fine[f01 + c]) +
              wyf         * ((1.f - wxf) * fine[f10 + c] + wxf * fine[f11 + c]);
    frow[NCLS + c] = (_Float16)v;
  }

  // coarse grid 128x128
  float yc = cy * (float)CO - 0.5f, xc = cx * (float)CO - 0.5f;
  float yc0f = floorf(yc), xc0f = floorf(xc);
  float wyc = yc - yc0f, wxc = xc - xc0f;
  int cy0 = min(max((int)yc0f, 0), CO - 1), cy1 = min(max((int)yc0f + 1, 0), CO - 1);
  int cx0 = min(max((int)xc0f, 0), CO - 1), cx1 = min(max((int)xc0f + 1, 0), CO - 1);

  if (lane < NCLS) {
    int c = lane;
    const float* c00 = coarse + (((size_t)b * CO + cy0) * CO + cx0) * NCLS;
    const float* c01 = coarse + (((size_t)b * CO + cy0) * CO + cx1) * NCLS;
    const float* c10 = coarse + (((size_t)b * CO + cy1) * CO + cx0) * NCLS;
    const float* c11 = coarse + (((size_t)b * CO + cy1) * CO + cx1) * NCLS;
    float v = (1.f - wyc) * ((1.f - wxc) * c00[c] + wxc * c01[c]) +
              wyc         * ((1.f - wxc) * c10[c] + wxc * c11[c]);
    _Float16 h = (_Float16)v;
    frow[c] = h;                                    // layer0 input cols 0..18
    X[(size_t)pt * KF + HID + c] = h;               // residual concat cols 256..274
  } else {
    int col = HID + lane;                           // lanes 19..31 -> cols 275..287 (pad)
    frow[col] = (_Float16)0.f;
    X[(size_t)pt * KF + col] = (_Float16)0.f;
  }
}

// ---------------- 5) weight -> WMMA B-fragment layout ----------------
// fragment half index = ((kc*NT + nt)*32 + lane)*16 + e ; B K-map: k = kc*32 + (e<8 ? 8*(lane>>4)+e : 16+8*(lane>>4)+e-8)
__global__ void __launch_bounds__(256) k_prepw(const float* __restrict__ W, int Krows, int Ncols,
                                               int NT, _Float16* __restrict__ dst) {
  int i = blockIdx.x * blockDim.x + threadIdx.x;
  int total = 9 * NT * 512;
  if (i >= total) return;
  int e = i & 15, lane = (i >> 4) & 31;
  int rest = i >> 9;
  int nt = rest % NT, kc = rest / NT;
  int g = lane >> 4, nl = lane & 15;
  int k = kc * 32 + ((e < 8) ? (8 * g + e) : (16 + 8 * g + (e - 8)));
  int n = nt * 16 + nl;
  float v = (k < Krows && n < Ncols) ? W[(size_t)k * Ncols + n] : 0.f;
  dst[i] = (_Float16)v;
}

// A-fragment K offset within a 32-chunk, per ISA 16-bit A 16x32 layout
__device__ __forceinline__ int akoff(int v, int g) {
  return (v < 4) ? (8 * g + 2 * v) : (16 + 8 * g + 2 * (v - 4));
}

// ---------------- 6) hidden MLP layer: Y[:,0:256] = relu(X @ W + b), f16, in-place safe ----------------
__global__ void __launch_bounds__(256) k_mlp_hidden(const _Float16* __restrict__ Xin,
                                                    const _Float16* __restrict__ Wfrag,
                                                    const float* __restrict__ bias,
                                                    _Float16* __restrict__ Xout) {
  int tid = threadIdx.x;
  int wave = tid >> 5, lane = tid & 31;
  int mrow = wave & 1, ncol = wave >> 1;            // 2 M-tiles x 4 N-groups(64) per block
  int g = lane >> 4, nl = lane & 15;
  int mBase = blockIdx.x * 32;
  int arow = mBase + mrow * 16 + nl;
  const _Float16* rowPtr = Xin + (size_t)arow * KF;

  v8f acc[4];
#pragma unroll
  for (int nt = 0; nt < 4; ++nt) acc[nt] = (v8f){0.f,0.f,0.f,0.f,0.f,0.f,0.f,0.f};

  for (int kc = 0; kc < 9; ++kc) {
    union { unsigned u[8]; v16h v; } a;
#pragma unroll
    for (int v = 0; v < 8; ++v)
      a.u[v] = *(const unsigned*)(rowPtr + kc * 32 + akoff(v, g));
    if (kc < 8)   // warm next chunk's B fragments
      __builtin_prefetch(Wfrag + ((size_t)((kc + 1) * 16 + ncol * 4) * 32 + lane) * 16, 0, 1);
#pragma unroll
    for (int nt = 0; nt < 4; ++nt) {
      const v16h* bp = (const v16h*)(Wfrag + ((size_t)(kc * 16 + ncol * 4 + nt) * 32 + lane) * 16);
      v16h bf = *bp;
      acc[nt] = __builtin_amdgcn_wmma_f32_16x16x32_f16(false, a.v, false, bf,
                                                       (short)0, acc[nt], false, false);
    }
  }

  __syncthreads();  // all A rows read before any in-place writeback

#pragma unroll
  for (int nt = 0; nt < 4; ++nt) {
    int n = ncol * 64 + nt * 16 + nl;
    float bv = bias[n];
#pragma unroll
    for (int r = 0; r < 8; ++r) {
      int m = mBase + mrow * 16 + 8 * g + r;        // f32 C layout: M = 8*laneGroup + vgpr
      float v = acc[nt][r] + bv;
      v = v > 0.f ? v : 0.f;
      Xout[(size_t)m * KF + n] = (_Float16)v;
    }
  }
}

// ---------------- 7) final layer: logits = X @ wf + bf (f32 out, no relu) ----------------
__global__ void __launch_bounds__(256) k_mlp_final(const _Float16* __restrict__ Xin,
                                                   const _Float16* __restrict__ Wfrag,
                                                   const float* __restrict__ bias,
                                                   float* __restrict__ out) {
  int tid = threadIdx.x;
  int wave = tid >> 5, lane = tid & 31;
  int mrow = wave >> 1, ncol = wave & 1;            // 4 M-tiles x 2 N-tiles per block (M=64)
  int g = lane >> 4, nl = lane & 15;
  int mBase = blockIdx.x * 64;
  int arow = mBase + mrow * 16 + nl;
  const _Float16* rowPtr = Xin + (size_t)arow * KF;

  v8f acc = (v8f){0.f,0.f,0.f,0.f,0.f,0.f,0.f,0.f};
  for (int kc = 0; kc < 9; ++kc) {
    union { unsigned u[8]; v16h v; } a;
#pragma unroll
    for (int v = 0; v < 8; ++v)
      a.u[v] = *(const unsigned*)(rowPtr + kc * 32 + akoff(v, g));
    const v16h* bp = (const v16h*)(Wfrag + ((size_t)(kc * 2 + ncol) * 32 + lane) * 16);
    v16h bf = *bp;
    acc = __builtin_amdgcn_wmma_f32_16x16x32_f16(false, a.v, false, bf,
                                                 (short)0, acc, false, false);
  }

  int n = ncol * 16 + nl;
  if (n < NCLS) {
    float bv = bias[n];
#pragma unroll
    for (int r = 0; r < 8; ++r) {
      int m = mBase + mrow * 16 + 8 * g + r;
      out[(size_t)m * NCLS + n] = acc[r] + bv;
    }
  }
}

// ---------------- 8) bilinear upsample 128->512 + softmax ----------------
__global__ void __launch_bounds__(256) k_upsoft(const float* __restrict__ coarse,
                                                float* __restrict__ out) {
  int i = blockIdx.x * blockDim.x + threadIdx.x;
  if (i >= NB * 512 * 512) return;
  int b = i >> 18, rem = i & (512 * 512 - 1);
  int y = rem >> 9, x = rem & 511;
  float sy = ((float)y + 0.5f) * 0.25f - 0.5f;
  float sx = ((float)x + 0.5f) * 0.25f - 0.5f;
  float y0f = floorf(sy), x0f = floorf(sx);
  float wy = sy - y0f, wx = sx - x0f;
  int y0 = min(max((int)y0f, 0), CO - 1), y1 = min(max((int)y0f + 1, 0), CO - 1);
  int x0 = min(max((int)x0f, 0), CO - 1), x1 = min(max((int)x0f + 1, 0), CO - 1);
  const float* f00 = coarse + (((size_t)b * CO + y0) * CO + x0) * NCLS;
  const float* f01 = coarse + (((size_t)b * CO + y0) * CO + x1) * NCLS;
  const float* f10 = coarse + (((size_t)b * CO + y1) * CO + x0) * NCLS;
  const float* f11 = coarse + (((size_t)b * CO + y1) * CO + x1) * NCLS;
  float l[NCLS];
  float m = -3.4e38f;
#pragma unroll
  for (int c = 0; c < NCLS; ++c) {
    float v = (1.f - wy) * ((1.f - wx) * f00[c] + wx * f01[c]) +
              wy         * ((1.f - wx) * f10[c] + wx * f11[c]);
    l[c] = v;
    m = v > m ? v : m;
  }
  float s = 0.f;
#pragma unroll
  for (int c = 0; c < NCLS; ++c) { float e = __expf(l[c] - m); l[c] = e; s += e; }
  float inv = 1.f / s;
  float* o = out + (size_t)i * NCLS;
#pragma unroll
  for (int c = 0; c < NCLS; ++c) o[c] = l[c] * inv;
}

// ---------------- launch ----------------
extern "C" void kernel_launch(void* const* d_in, const int* in_sizes, int n_in,
                              void* d_out, int out_size, void* d_ws, size_t ws_size,
                              hipStream_t stream) {
  (void)in_sizes; (void)n_in; (void)out_size; (void)ws_size;
  const float* coarse = (const float*)d_in[1];
  const float* fine0  = (const float*)d_in[2];
  const float* rc     = (const float*)d_in[3];
  const float* rf     = (const float*)d_in[4];
  const float* w0 = (const float*)d_in[5];  const float* b0 = (const float*)d_in[6];
  const float* w1 = (const float*)d_in[7];  const float* b1 = (const float*)d_in[8];
  const float* w2 = (const float*)d_in[9];  const float* b2 = (const float*)d_in[10];
  const float* wf = (const float*)d_in[11]; const float* bf = (const float*)d_in[12];

  float* out = (float*)d_out;
  char*  ws  = (char*)d_ws;
  unsigned long long* keys = (unsigned long long*)(ws + WS_KEYS);
  _Float16* wfrag0 = (_Float16*)(ws + WS_WF0);
  _Float16* wfrag1 = (_Float16*)(ws + WS_WF1);
  _Float16* wfrag2 = (_Float16*)(ws + WS_WF2);
  _Float16* wfragf = (_Float16*)(ws + WS_WFF);
  _Float16* feat0  = (_Float16*)(ws + WS_FEAT0);
  _Float16* X      = (_Float16*)(ws + WS_X);

  // weights -> WMMA fragment layout
  k_prepw<<<(9 * 16 * 512 + 255) / 256, 256, 0, stream>>>(w0, NCLS + FIC, HID, 16, wfrag0);
  k_prepw<<<(9 * 16 * 512 + 255) / 256, 256, 0, stream>>>(w1, HID + NCLS, HID, 16, wfrag1);
  k_prepw<<<(9 * 16 * 512 + 255) / 256, 256, 0, stream>>>(w2, HID + NCLS, HID, 16, wfrag2);
  k_prepw<<<(9 * 2 * 512 + 255) / 256, 256, 0, stream>>>(wf, HID + NCLS, NCLS, 2, wfragf);

  // uncertainty + exact sorted top-k (global bitonic, descending)
  k_uncertainty<<<(NB * NPAD) / 256, 256, 0, stream>>>(coarse, rc, keys);
  for (int k = 2; k <= NPAD; k <<= 1)
    for (int j = k >> 1; j >= 1; j >>= 1)
      k_sortstep<<<(NB * NPAD) / 256, 256, 0, stream>>>(keys, j, k);

  // coords, features
  k_gather<<<NPTS / 256, 256, 0, stream>>>(keys, rc, rf, out + OUT_COORDS);
  k_sample<<<NPTS / 8, 256, 0, stream>>>(coarse, fine0, out + OUT_COORDS, feat0, X);

  // point head: 3 hidden layers (in-place on X) + final
  k_mlp_hidden<<<NPTS / 32, 256, 0, stream>>>(feat0, wfrag0, b0, X);
  k_mlp_hidden<<<NPTS / 32, 256, 0, stream>>>(X,     wfrag1, b1, X);
  k_mlp_hidden<<<NPTS / 32, 256, 0, stream>>>(X,     wfrag2, b2, X);
  k_mlp_final <<<NPTS / 64, 256, 0, stream>>>(X,     wfragf, bf, out + OUT_LOGITS);

  // upsample + softmax (independent)
  k_upsoft<<<(NB * 512 * 512) / 256, 256, 0, stream>>>(coarse, out);
}
